// LDConv_MultiScale_70093866270952
// MI455X (gfx1250) — matrix operand order repaired and checked by reading, not compiled, not run on hardware
//
#include <hip/hip_runtime.h>
#include <hip/hip_bf16.h>

typedef __attribute__((ext_vector_type(16))) _Float16 v16h;
typedef __attribute__((ext_vector_type(8)))  float    v8f;

union FragH { v16h v; uint4 q[2]; };

constexpr int Hh   = 128;
constexpr int Ww   = 128;
constexpr int HW   = Hh * Ww;          // 16384
constexpr int NB   = 8;                // batch
constexpr int NPIX = NB * HW;          // 131072
constexpr int KC   = 576;              // im2col K = 9 taps * 64 ch
constexpr int HP   = 130;              // padded spatial (1-px halo)

// workspace byte offsets (all 256-aligned); total ~185.2 MB
constexpr unsigned long long XH_OFF     = 0ull;          // NHWC f16 x, padded 8x130x130x64 : 17.31 MB
constexpr unsigned long long XF_OFF     = 17305600ull;   // NHWC f32 x : 33.55 MB
constexpr unsigned long long BMAT_OFF   = 50860032ull;   // conv weights f16 [576][64]
constexpr unsigned long long BC_OFF     = 50933760ull;   // mix weights f16 [3][64][64]
constexpr unsigned long long STATS_OFF  = 50958336ull;   // f32 [3][128] (sum, sumsq)
constexpr unsigned long long OFFRAW_OFF = 50959872ull;   // f32 [NPIX][64] : 33.55 MB
constexpr unsigned long long Y_OFF      = 84514304ull;   // f32 [NPIX][3][64] : 100.7 MB

constexpr int XH_U4 = 17305600 / 16;   // uint4 count of padded xh region

// ---------------- phase 0a0: zero padded xh (provides the halo) ------------
__global__ void __launch_bounds__(256)
zero_xh(uint4* __restrict__ p) {
  int idx = blockIdx.x * 256 + threadIdx.x;
  p[idx] = make_uint4(0u, 0u, 0u, 0u);
}

// ---------------- phase 0a: x -> NHWC (f16 padded for conv A, f32 gather) --
__global__ void __launch_bounds__(256)
prep_x(const float* __restrict__ x, _Float16* __restrict__ xh, float* __restrict__ xf) {
  int idx = blockIdx.x * 256 + threadIdx.x;            // NCHW linear
  float v = x[idx];
  int j = idx & 127, i = (idx >> 7) & 127, c = (idx >> 14) & 63, b = idx >> 20;
  xf[(((b << 14) | (i << 7) | j) << 6) | c] = v;       // unpadded NHWC f32
  long ph = ((((long)b * HP + (i + 1)) * HP + (j + 1)) << 6) + c;
  xh[ph] = (_Float16)v;                                // padded NHWC f16
}

// ---------------- phase 0b: weight repack + stats zero ---------------------
__global__ void __launch_bounds__(256)
prep_w(const float* __restrict__ Wp, const float* __restrict__ Wc,
       _Float16* __restrict__ Bmat, _Float16* __restrict__ Bc,
       float* __restrict__ stats) {
  int idx = blockIdx.x * 256 + threadIdx.x;
  if (idx < KC * 64) {                       // conv B: [k=tap*64+c][n=s*18+o], pad n>=54 with 0
    int k = idx >> 6, n = idx & 63;
    int tap = k >> 6, c = k & 63;
    _Float16 v = (_Float16)0.f;
    if (n < 54) {
      int s = n / 18, o = n % 18;            // Wp shape (3,18,64,3,3)
      v = (_Float16)Wp[((s * 18 + o) * 64 + c) * 9 + tap];
    }
    Bmat[idx] = v;
  } else if (idx < KC * 64 + 3 * 64 * 64) {  // mix B: Bc[s][c][o] = Wc[s][o][c]
    int t = idx - KC * 64;
    int s = t >> 12, c = (t >> 6) & 63, o = t & 63;
    Bc[t] = (_Float16)Wc[(s << 12) + (o << 6) + c];
  } else if (idx < KC * 64 + 3 * 64 * 64 + 384) {
    stats[idx - (KC * 64 + 3 * 64 * 64)] = 0.f;
  }
}

// ---------------- phase 1: fused 3-scale offset conv as WMMA GEMM ----------
// M=NPIX, K=576, N=64 (54 real). One wave: 16 pixels x 64 outputs.
// A loads are unconditional (halo-padded xh). B staged in LDS in two K-halves
// of 288 rows (41.5 KB, padded pitch 72 halfs to break bank-conflict stride).
__global__ void __launch_bounds__(256)
offset_gemm(const _Float16* __restrict__ xh, const _Float16* __restrict__ Bm,
            float* __restrict__ offr) {
  __shared__ __align__(16) _Float16 Bl[288 * 72];      // 41472 B
  const int tid  = threadIdx.x;
  const int lane = tid & 31;
  const int wave = tid >> 5;
  const int pixw = (blockIdx.x * 8 + wave) * 16;
  const int m = lane & 15, g = lane >> 4;
  const int pix = pixw + m;
  const int b = pix >> 14, rem = pix & 16383, i = rem >> 7, j = rem & 127;
  const long rowbase = ((long)b * HP + i) * HP + j;    // halo cancels the -1

  v8f acc0 = {}, acc1 = {}, acc2 = {}, acc3 = {};
  for (int half = 0; half < 2; ++half) {
    __syncthreads();                                   // previous half fully consumed
    for (int c = tid; c < 288 * 8; c += 256) {         // stage 288 K-rows of B
      int row = c >> 3, part = c & 7;
      *(uint4*)(Bl + row * 72 + part * 8) =
          *(const uint4*)(Bm + (half * 288 + row) * 64 + part * 8);
    }
    __syncthreads();
    for (int t = 0; t < 9; ++t) {
      const int kl = t * 32;                           // K base within this half
      const int k0 = half * 288 + kl;                  // global K base
      FragH A;
#pragma unroll
      for (int e = 0; e < 2; ++e) {                    // two 8-half K-octets per lane
        int ks = k0 + e * 16 + g * 8;                  // octet never crosses a tap
        int tap = ks >> 6, coff = ks & 63;
        A.q[e] = *(const uint4*)(xh + ((rowbase + (tap / 3) * HP + tap % 3) << 6) + coff);
      }
      FragH B0, B1, B2, B3;
#pragma unroll
      for (int e = 0; e < 16; ++e) {
        const _Float16* br = Bl + (kl + (e & 7) + g * 8 + (e >> 3) * 16) * 72 + m;
        B0.v[e] = br[0]; B1.v[e] = br[16]; B2.v[e] = br[32]; B3.v[e] = br[48];
      }
      acc0 = __builtin_amdgcn_wmma_f32_16x16x32_f16(false, A.v, false, B0.v, (short)0, acc0, false, false);
      acc1 = __builtin_amdgcn_wmma_f32_16x16x32_f16(false, A.v, false, B1.v, (short)0, acc1, false, false);
      acc2 = __builtin_amdgcn_wmma_f32_16x16x32_f16(false, A.v, false, B2.v, (short)0, acc2, false, false);
      acc3 = __builtin_amdgcn_wmma_f32_16x16x32_f16(false, A.v, false, B3.v, (short)0, acc3, false, false);
    }
  }
  const int n = lane & 15;
  const int mb = g * 8;                      // D: lane%16 = N, vgpr r -> M = r + 8*(lane/16)
#pragma unroll
  for (int r = 0; r < 8; ++r) {
    float* dst = offr + (long)(pixw + mb + r) * 64 + n;
    dst[0] = acc0[r]; dst[16] = acc1[r]; dst[32] = acc2[r]; dst[48] = acc3[r];
  }
}

// ---------------- phase 2: bilinear gather + WMMA channel mix + BN stats ---
// grid.y = scale. Block = 8 waves = 32 pixels. Wave gathers 4 pixels (lane=channel),
// stages x_off f16 in LDS, then each wave does one 16x16 tile of the 64x64 GEMM.
__global__ void __launch_bounds__(256)
gather_mix(const float* __restrict__ xf, const float* __restrict__ offr,
           const float* __restrict__ bp, const _Float16* __restrict__ Bc,
           float* __restrict__ yws, float* __restrict__ stats) {
  const int s = blockIdx.y;
  const float scalef = (float)(s + 1);       // SCALES = 1,2,3
  const int lane = threadIdx.x & 31;
  const int wave = threadIdx.x >> 5;
  const int pixbase = blockIdx.x * 32;

  __shared__ __align__(16) _Float16 Axo[32][64];
  __shared__ float red[2][64];
  if (threadIdx.x < 64) { red[0][threadIdx.x] = 0.f; red[1][threadIdx.x] = 0.f; }
  __syncthreads();

  const int c0 = lane, c1 = lane + 32;
  for (int q = 0; q < 4; ++q) {
    int pix = pixbase + wave * 4 + q;
    int b = pix >> 14, rem = pix & 16383, i = rem >> 7, j = rem & 127;
    const float* op  = offr + (long)pix * 64 + s * 18;
    const float* bps = bp + s * 18;
    const float* xb  = xf + (long)b * HW * 64;
    float a0 = 0.f, a1 = 0.f;
#pragma unroll
    for (int n = 0; n < 9; ++n) {
      float px = (float)i + (float)(n / 3) + (op[n]     + bps[n])     * scalef;
      float py = (float)j + (float)(n % 3) + (op[n + 9] + bps[n + 9]) * scalef;
      float fx = floorf(px), fy = floorf(py);
      int qx = (int)fx, qy = (int)fy;
      float dx = px - fx, dy = py - fy;
      float glt = (1.f - dx) * (1.f - dy), grb = dx * dy;
      float glb = (1.f - dx) * dy,         grt = dx * (1.f - dy);
      int flt = qx * Ww + qy;               // reference clamps the FLAT index
      int f_lt = min(max(flt,          0), HW - 1);
      int f_rb = min(max(flt + Ww + 1, 0), HW - 1);
      int f_lb = min(max(flt + 1,      0), HW - 1);
      int f_rt = min(max(flt + Ww,     0), HW - 1);
      const float* plt = xb + (long)f_lt * 64;
      const float* prb = xb + (long)f_rb * 64;
      const float* plb = xb + (long)f_lb * 64;
      const float* prt = xb + (long)f_rt * 64;
      a0 += glt * plt[c0] + grb * prb[c0] + glb * plb[c0] + grt * prt[c0];
      a1 += glt * plt[c1] + grb * prb[c1] + glb * plb[c1] + grt * prt[c1];
    }
    Axo[wave * 4 + q][c0] = (_Float16)(a0 * (1.f / 9.f));
    Axo[wave * 4 + q][c1] = (_Float16)(a1 * (1.f / 9.f));
  }
  __syncthreads();

  const int mt = wave >> 2, nt = wave & 3;   // 2 M-tiles x 4 N-tiles
  const int m = lane & 15, g = lane >> 4;
  v8f acc = {};
#pragma unroll
  for (int kt = 0; kt < 2; ++kt) {
    FragH A, B;
    int ks = kt * 32;
    A.q[0] = *(const uint4*)&Axo[mt * 16 + m][ks + g * 8];
    A.q[1] = *(const uint4*)&Axo[mt * 16 + m][ks + 16 + g * 8];
#pragma unroll
    for (int e = 0; e < 16; ++e)
      B.v[e] = Bc[(s << 12) + (ks + (e & 7) + g * 8 + (e >> 3) * 16) * 64 + nt * 16 + m];
    acc = __builtin_amdgcn_wmma_f32_16x16x32_f16(false, A.v, false, B.v, (short)0, acc, false, false);
  }
  const int n = nt * 16 + m;
  float s1 = 0.f, s2 = 0.f;
#pragma unroll
  for (int r = 0; r < 8; ++r) {
    int mm = mt * 16 + g * 8 + r;
    float v = acc[r];
    yws[(long)(pixbase + mm) * 192 + s * 64 + n] = v;
    s1 += v; s2 += v * v;
  }
  atomicAdd(&red[0][n], s1);                 // LDS ds_add_f32 reduce
  atomicAdd(&red[1][n], s2);
  __syncthreads();
  if (threadIdx.x < 64) {
    atomicAdd(&stats[s * 128 + threadIdx.x],      red[0][threadIdx.x]);
    atomicAdd(&stats[s * 128 + 64 + threadIdx.x], red[1][threadIdx.x]);
  }
}

// ---------------- phase 3: BN normalize + SiLU + scale-average -------------
__global__ void __launch_bounds__(256)
finalize(const float* __restrict__ yws, const float* __restrict__ stats,
         const float* __restrict__ gamma, const float* __restrict__ beta,
         float* __restrict__ out) {
  int idx = blockIdx.x * 256 + threadIdx.x;  // NCHW linear output
  int j = idx & 127, i = (idx >> 7) & 127, o = (idx >> 14) & 63, b = idx >> 20;
  int pix = (b << 14) | (i << 7) | j;
  const float inv = 1.f / (float)NPIX;
  float acc = 0.f;
#pragma unroll
  for (int s = 0; s < 3; ++s) {
    float mu  = stats[s * 128 + o] * inv;
    float var = stats[s * 128 + 64 + o] * inv - mu * mu;
    float y   = yws[(long)pix * 192 + s * 64 + o];
    float yn  = (y - mu) * rsqrtf(var + 1e-5f) * gamma[s * 64 + o] + beta[s * 64 + o];
    acc += yn * (1.f / (1.f + __expf(-yn)));
  }
  out[idx] = acc * (1.f / 3.f);
}

extern "C" void kernel_launch(void* const* d_in, const int* in_sizes, int n_in,
                              void* d_out, int out_size, void* d_ws, size_t ws_size,
                              hipStream_t stream) {
  const float* x     = (const float*)d_in[0];
  const float* Wp    = (const float*)d_in[1];
  const float* bp    = (const float*)d_in[2];
  const float* Wc    = (const float*)d_in[3];
  const float* gamma = (const float*)d_in[4];
  const float* beta  = (const float*)d_in[5];
  float* out = (float*)d_out;

  char* wsb = (char*)d_ws;
  _Float16* xh   = (_Float16*)(wsb + XH_OFF);
  float*    xf   = (float*)   (wsb + XF_OFF);
  _Float16* Bmat = (_Float16*)(wsb + BMAT_OFF);
  _Float16* Bc   = (_Float16*)(wsb + BC_OFF);
  float*    st   = (float*)   (wsb + STATS_OFF);
  float*    offr = (float*)   (wsb + OFFRAW_OFF);
  float*    yws  = (float*)   (wsb + Y_OFF);

  zero_xh<<<XH_U4 / 256, 256, 0, stream>>>((uint4*)xh);
  prep_x<<<NB * 64 * HW / 256, 256, 0, stream>>>(x, xh, xf);
  prep_w<<<(KC * 64 + 3 * 64 * 64 + 384 + 255) / 256, 256, 0, stream>>>(Wp, Wc, Bmat, Bc, st);
  offset_gemm<<<NPIX / 128, 256, 0, stream>>>(xh, Bmat, offr);
  dim3 gg(NPIX / 32, 3, 1);
  gather_mix<<<gg, 256, 0, stream>>>(xf, offr, bp, Bc, yws, st);
  finalize<<<NB * 64 * HW / 256, 256, 0, stream>>>(yws, st, gamma, beta, out);
}
